// TransformerBlock_90898687852801
// MI455X (gfx1250) — matrix operand order repaired and checked by reading, not compiled, not run on hardware
//
#include <hip/hip_runtime.h>

// ---------------------------------------------------------------------------
// Problem constants (from the reference)
// ---------------------------------------------------------------------------
static constexpr int Bc = 4, Cch = 256, Hc = 24, Wc = 24, Dc = 24;
static constexpr int NHc = 8, HDc = 32;
static constexpr int NT = Hc * Wc * Dc;          // 13824 tokens per batch
static constexpr int BT = Bc * NT;               // 55296 rows
static constexpr int KCONV = 27 * 256;           // 6912

typedef unsigned short u16;
typedef __attribute__((ext_vector_type(16))) __bf16 v16bf;
typedef __attribute__((ext_vector_type(8)))  float  v8f;

// ---------------------------------------------------------------------------
// Workspace arena (bytes). All offsets 256-aligned by construction.
// ---------------------------------------------------------------------------
static constexpr size_t SZ_F32T  = (size_t)BT * 256 * 4;       // 56,623,104
static constexpr size_t SZ_BF16T = (size_t)BT * 256 * 2;       // 28,311,552
static constexpr size_t O_QKVV = 0;                            // BT x 1024 f32
static constexpr size_t O_XN   = O_QKVV + (size_t)BT * 1024 * 4;
static constexpr size_t O_QS   = O_XN  + SZ_BF16T;             // (b,h,t,e) f32
static constexpr size_t O_KN   = O_QS  + SZ_F32T;
static constexpr size_t O_XSA  = O_KN  + SZ_F32T;              // bf16
static constexpr size_t O_XCA  = O_XSA + SZ_BF16T;             // bf16
static constexpr size_t O_EPA  = O_XCA + SZ_BF16T;             // f32
static constexpr size_t O_SKIP = O_EPA + SZ_F32T;              // f32 NCDHW
static constexpr size_t O_W1   = O_SKIP + SZ_F32T;             // bf16 [co][27*256]
static constexpr size_t O_W2   = O_W1 + (size_t)256 * KCONV * 2;
static constexpr size_t O_WQ   = O_W2 + (size_t)256 * KCONV * 2;
static constexpr size_t O_WOP  = O_WQ   + (size_t)1024 * 256 * 2;
static constexpr size_t O_WOP2 = O_WOP  + (size_t)128 * 256 * 2;
static constexpr size_t O_W8   = O_WOP2 + (size_t)128 * 256 * 2;
static constexpr size_t O_STATS= O_W8   + (size_t)256 * 256 * 2; // 512 f32
static constexpr size_t O_QCN  = O_STATS + 512 * 4;              // 1024 f32
static constexpr size_t O_KCN  = O_QCN + 1024 * 4;
static constexpr size_t O_ATTN = O_KCN + 1024 * 4;               // 32*32*32 f32
// Aliases into the QKVV region (dead after attention phase):
static constexpr size_t O_CONVOUT = 0;                 // f32 channel-last
static constexpr size_t O_ACT     = 64ull << 20;       // bf16 channel-last
static constexpr size_t O_O8      = 96ull << 20;       // f32 channel-last

// ---------------------------------------------------------------------------
// Small helpers
// ---------------------------------------------------------------------------
__device__ inline u16 f2bf(float f) {
  union { float f; unsigned u; } v; v.f = f;
  unsigned r = v.u + 0x7FFFu + ((v.u >> 16) & 1u);
  return (u16)(r >> 16);
}
__device__ inline float wsum(float v) {
#pragma unroll
  for (int m = 16; m; m >>= 1) v += __shfl_xor(v, m, 32);
  return v;
}
__device__ inline float wmax(float v) {
#pragma unroll
  for (int m = 16; m; m >>= 1) v = fmaxf(v, __shfl_xor(v, m, 32));
  return v;
}

// ---------------------------------------------------------------------------
// WMMA fragment helpers.  LDS tiles are 128 rows x 32 bf16, row stride padded
// to 40 ushorts (80B): 20*m mod 64 banks are spaced exactly 4 apart ->
// conflict-free ds_read_b128 across 16 lanes.
// ---------------------------------------------------------------------------
static constexpr int LDP = 40;
struct Frag { union { v16bf v; uint4 q[2]; }; };

// A fragment (16x32 bf16): lane<16 -> row m0+lane, K {0..7,16..23};
// lane>=16 -> row m0+lane-16, K {8..15,24..31}.
__device__ inline void lda_frag(Frag& f, const u16* lds, int m0) {
  const int lane = threadIdx.x & 31;
  const u16* p = lds + (m0 + (lane & 15)) * LDP + ((lane >> 4) << 3);
  f.q[0] = *(const uint4*)(p);
  f.q[1] = *(const uint4*)(p + 16);
}
// B fragment (32x16 bf16), tile stored [n][k]: lane<16 -> col n0+lane K {0..7,16..23},
// lane>=16 -> col n0+lane-16 K {8..15,24..31}.
__device__ inline void ldb_frag(Frag& f, const u16* lds, int n0) {
  const int lane = threadIdx.x & 31;
  const u16* p = lds + (n0 + (lane & 15)) * LDP + ((lane >> 4) << 3);
  f.q[0] = *(const uint4*)(p);
  f.q[1] = *(const uint4*)(p + 16);
}

// ---------------------------------------------------------------------------
// Generic bf16 WMMA GEMM: C[M x N](f32) = A[M x K](bf16) * B[N x K]^T(bf16)
// BM=BN=128, BK=32; 256 threads = 8 waves, each wave 64x32 (4x2 fragments).
// ---------------------------------------------------------------------------
__global__ __launch_bounds__(256)
void gemm_bf16(const u16* __restrict__ A, const u16* __restrict__ Bw,
               float* __restrict__ Cout, int M, int N, int K,
               int ldc, int colOff, const float* __restrict__ bias)
{
  __shared__ u16 sA[128 * LDP];
  __shared__ u16 sB[128 * LDP];
  const int NTl = N >> 7;
  const int M0 = (blockIdx.x / NTl) << 7;
  const int N0 = (blockIdx.x % NTl) << 7;
  const int tid  = threadIdx.x;
  const int r    = tid >> 1;
  const int colc = (tid & 1) << 4;
  const int wave = tid >> 5;
  const int wm = (wave >> 2) * 64;
  const int wn = (wave & 3) * 32;

  const v8f vzero = {};
  v8f acc[4][2];
#pragma unroll
  for (int i = 0; i < 4; ++i)
#pragma unroll
    for (int j = 0; j < 2; ++j) acc[i][j] = vzero;

  for (int k0 = 0; k0 < K; k0 += 32) {
    const u16* ga = A  + (size_t)(M0 + r) * K + k0 + colc;
    const u16* gb = Bw + (size_t)(N0 + r) * K + k0 + colc;
    *(uint4*)&sA[r * LDP + colc]     = *(const uint4*)(ga);
    *(uint4*)&sA[r * LDP + colc + 8] = *(const uint4*)(ga + 8);
    *(uint4*)&sB[r * LDP + colc]     = *(const uint4*)(gb);
    *(uint4*)&sB[r * LDP + colc + 8] = *(const uint4*)(gb + 8);
    __syncthreads();
    Frag af[4], bfr[2];
#pragma unroll
    for (int i = 0; i < 4; ++i) lda_frag(af[i], sA, wm + 16 * i);
#pragma unroll
    for (int j = 0; j < 2; ++j) ldb_frag(bfr[j], sB, wn + 16 * j);
#pragma unroll
    for (int i = 0; i < 4; ++i)
#pragma unroll
      for (int j = 0; j < 2; ++j)
        acc[i][j] = __builtin_amdgcn_wmma_f32_16x16x32_bf16(
            false, af[i].v, false, bfr[j].v, (short)0, acc[i][j], false, false);
    __syncthreads();
  }

  const int lane  = threadIdx.x & 31;
  const int colL  = lane & 15;
  const int rowHi = (lane >> 4) << 3;
#pragma unroll
  for (int i = 0; i < 4; ++i)
#pragma unroll
    for (int j = 0; j < 2; ++j) {
      const int col = N0 + wn + 16 * j + colL;
      const float bv = bias ? bias[col] : 0.f;
#pragma unroll
      for (int rr = 0; rr < 8; ++rr) {
        const int row = M0 + wm + 16 * i + rowHi + rr;
        Cout[(size_t)row * ldc + colOff + col] = acc[i][j][rr] + bv;
      }
    }
}

// ---------------------------------------------------------------------------
// Implicit-GEMM 3x3x3 conv (C=256->256, SAME) via WMMA.
// act: bf16 channel-last (b,t,ci); wr: bf16 [co][tap*256+ci]; out f32 (b,t,co)
// ---------------------------------------------------------------------------
__global__ __launch_bounds__(256)
void conv3d_bf16(const u16* __restrict__ act, const u16* __restrict__ wr,
                 float* __restrict__ out)
{
  __shared__ u16 sA[128 * LDP];
  __shared__ u16 sB[128 * LDP];
  const int mtile = blockIdx.x >> 1;       // 432 M-tiles (128 tokens each)
  const int N0    = (blockIdx.x & 1) << 7;
  const int bb = (mtile * 128) / NT;
  const int t0 = (mtile * 128) % NT;       // 13824 = 108*128: tile in-batch
  const int tid  = threadIdx.x;
  const int r    = tid >> 1;
  const int colc = (tid & 1) << 4;
  const int tloc = t0 + r;
  const int hh = tloc / (Wc * Dc), ww = (tloc / Dc) % Wc, dd = tloc % Dc;
  const int wave = tid >> 5;
  const int wm = (wave >> 2) * 64, wn = (wave & 3) * 32;

  const v8f vzero = {};
  v8f acc[4][2];
#pragma unroll
  for (int i = 0; i < 4; ++i)
#pragma unroll
    for (int j = 0; j < 2; ++j) acc[i][j] = vzero;

  for (int tap = 0; tap < 27; ++tap) {
    const int dh = tap / 9 - 1, dw = (tap / 3) % 3 - 1, dz = tap % 3 - 1;
    const bool valid = ((unsigned)(hh + dh) < (unsigned)Hc) &&
                       ((unsigned)(ww + dw) < (unsigned)Wc) &&
                       ((unsigned)(dd + dz) < (unsigned)Dc);
    const int ntok = tloc + dh * (Wc * Dc) + dw * Dc + dz;
    const u16* gaBase = act + ((size_t)(bb * NT + ntok)) * 256 + colc;
    const u16* gbBase = wr + (size_t)(N0 + r) * KCONV + tap * 256 + colc;
    for (int kc = 0; kc < 256; kc += 32) {
      uint4 a0 = {0, 0, 0, 0}, a1 = {0, 0, 0, 0};
      if (valid) { a0 = *(const uint4*)(gaBase + kc); a1 = *(const uint4*)(gaBase + kc + 8); }
      *(uint4*)&sA[r * LDP + colc]     = a0;
      *(uint4*)&sA[r * LDP + colc + 8] = a1;
      *(uint4*)&sB[r * LDP + colc]     = *(const uint4*)(gbBase + kc);
      *(uint4*)&sB[r * LDP + colc + 8] = *(const uint4*)(gbBase + kc + 8);
      __syncthreads();
      Frag af[4], bfr[2];
#pragma unroll
      for (int i = 0; i < 4; ++i) lda_frag(af[i], sA, wm + 16 * i);
#pragma unroll
      for (int j = 0; j < 2; ++j) ldb_frag(bfr[j], sB, wn + 16 * j);
#pragma unroll
      for (int i = 0; i < 4; ++i)
#pragma unroll
        for (int j = 0; j < 2; ++j)
          acc[i][j] = __builtin_amdgcn_wmma_f32_16x16x32_bf16(
              false, af[i].v, false, bfr[j].v, (short)0, acc[i][j], false, false);
      __syncthreads();
    }
  }

  const int lane  = threadIdx.x & 31;
  const int colL  = lane & 15;
  const int rowHi = (lane >> 4) << 3;
#pragma unroll
  for (int i = 0; i < 4; ++i)
#pragma unroll
    for (int j = 0; j < 2; ++j)
#pragma unroll
      for (int rr = 0; rr < 8; ++rr) {
        const int row = bb * NT + t0 + wm + 16 * i + rowHi + rr;
        out[(size_t)row * 256 + N0 + wn + 16 * j + colL] = acc[i][j][rr];
      }
}

// ---------------------------------------------------------------------------
// LayerNorm over C with NCDHW->token-major transpose through LDS.
// Tile: 32 tokens x 256 channels (pad 33 -> conflict-free both directions).
// ---------------------------------------------------------------------------
__global__ __launch_bounds__(256)
void ln_kernel(const float* __restrict__ x, const float* __restrict__ lw,
               const float* __restrict__ lb, u16* __restrict__ xn)
{
  __shared__ float ld[256 * 33];
  __shared__ float mu_s[32], rs_s[32];
  const int b  = blockIdx.x / (NT / 32);
  const int t0 = (blockIdx.x % (NT / 32)) * 32;
  const int tid = threadIdx.x;
  {
    const int j = tid & 31, cr = tid >> 5;
#pragma unroll
    for (int it = 0; it < 32; ++it) {
      const int c = it * 8 + cr;
      ld[c * 33 + j] = x[(size_t)(b * 256 + c) * NT + t0 + j];
    }
  }
  __syncthreads();
  if (tid < 32) {
    float s = 0.f, s2 = 0.f;
    for (int c = 0; c < 256; ++c) { const float v = ld[c * 33 + tid]; s += v; s2 += v * v; }
    const float mu = s * (1.f / 256.f);
    mu_s[tid] = mu;
    rs_s[tid] = rsqrtf(s2 * (1.f / 256.f) - mu * mu + 1e-5f);
  }
  __syncthreads();
  const float w = lw[tid], bo = lb[tid];
  for (int j = 0; j < 32; ++j) {
    const float v = (ld[tid * 33 + j] - mu_s[j]) * rs_s[j] * w + bo;
    xn[(size_t)(b * NT + t0 + j) * 256 + tid] = f2bf(v);
  }
}

// ---------------------------------------------------------------------------
// q/k L2-normalize per token (wave per (b,h,t), lane = hd element)
// ---------------------------------------------------------------------------
__global__ __launch_bounds__(256)
void qknorm_kernel(const float* __restrict__ qkvv, const float* __restrict__ qe,
                   const float* __restrict__ temp2, float* __restrict__ qs,
                   float* __restrict__ kn)
{
  const int lane = threadIdx.x & 31;
  const int job  = blockIdx.x * 8 + (threadIdx.x >> 5);
  const int b = job / (NHc * NT);
  const int rem = job % (NHc * NT);
  const int h = rem / NT, t = rem % NT;
  const size_t base = (size_t)(b * NT + t) * 1024 + h * 32 + lane;
  const float q = qkvv[base];
  const float k = qkvv[base + 256];
  const float qn2 = wsum(q * q), kn2 = wsum(k * k);
  const float qv = q / fmaxf(sqrtf(qn2), 1e-12f);
  const float kv = k / fmaxf(sqrtf(kn2), 1e-12f);
  const float tv = temp2[h];
  const float sp = (tv > 20.f) ? tv : log1pf(__expf(tv));
  qs[(size_t)job * 32 + lane] = (qv + qe[h * 32 + lane]) * sp;
  kn[(size_t)job * 32 + lane] = kv;
}

// ---------------------------------------------------------------------------
// 3x3x3 neighborhood attention: wave per (b,h,t); 27 dots via lane reductions
// ---------------------------------------------------------------------------
__global__ __launch_bounds__(256)
void sa_kernel(const float* __restrict__ qs, const float* __restrict__ kn,
               const float* __restrict__ qkvv, const float* __restrict__ rpb,
               u16* __restrict__ xsa)
{
  const int lane = threadIdx.x & 31;
  const int job  = blockIdx.x * 8 + (threadIdx.x >> 5);
  const int b = job / (NHc * NT);
  const int rem = job % (NHc * NT);
  const int h = rem / NT, t = rem % NT;
  const int hh = t / (Wc * Dc), ww = (t / Dc) % Wc, dd = t % Dc;
  const float qv = qs[(size_t)job * 32 + lane];
  const size_t knBase = (size_t)(b * NHc + h) * NT;
  float a[27];
#pragma unroll
  for (int idx = 0; idx < 27; ++idx) {
    const int dh = idx / 9 - 1, dw = (idx / 3) % 3 - 1, dz = idx % 3 - 1;
    const bool valid = ((unsigned)(hh + dh) < (unsigned)Hc) &&
                       ((unsigned)(ww + dw) < (unsigned)Wc) &&
                       ((unsigned)(dd + dz) < (unsigned)Dc);
    if (valid) {
      const int ntok = t + dh * (Wc * Dc) + dw * Dc + dz;
      const float kv = kn[(knBase + ntok) * 32 + lane];
      a[idx] = wsum(qv * kv) + rpb[h * 27 + idx];
    } else {
      a[idx] = -1e30f;
    }
  }
  float m = -1e30f;
#pragma unroll
  for (int idx = 0; idx < 27; ++idx) m = fmaxf(m, a[idx]);
  float s = 0.f;
#pragma unroll
  for (int idx = 0; idx < 27; ++idx) { a[idx] = __expf(a[idx] - m); s += a[idx]; }
  const float inv = 1.f / s;
  float out = 0.f;
#pragma unroll
  for (int idx = 0; idx < 27; ++idx) {
    const int dh = idx / 9 - 1, dw = (idx / 3) % 3 - 1, dz = idx % 3 - 1;
    const bool valid = ((unsigned)(hh + dh) < (unsigned)Hc) &&
                       ((unsigned)(ww + dw) < (unsigned)Wc) &&
                       ((unsigned)(dd + dz) < (unsigned)Dc);
    if (valid) {
      const int ntok = t + dh * (Wc * Dc) + dw * Dc + dz;
      out += a[idx] * inv *
             qkvv[(size_t)(b * NT + ntok) * 1024 + 768 + h * 32 + lane];
    }
  }
  xsa[(size_t)(b * NT + t) * 256 + h * 32 + lane] = f2bf(out);
}

// ---------------------------------------------------------------------------
// Channel attention: row norms over tokens, 32x32 attn, 32xN apply
// ---------------------------------------------------------------------------
__global__ __launch_bounds__(256)
void ca_norm_kernel(const float* __restrict__ qkvv, float* __restrict__ qcn,
                    float* __restrict__ kcn)
{
  __shared__ float sq[256], sk[256];
  const int b = blockIdx.x >> 3, h = blockIdx.x & 7;
  const int tt = threadIdx.x >> 5, c = threadIdx.x & 31;
  float aq = 0.f, ak = 0.f;
  for (int t = tt; t < NT; t += 8) {
    const size_t base = (size_t)(b * NT + t) * 1024 + h * 32 + c;
    const float q = qkvv[base], k = qkvv[base + 256];
    aq += q * q; ak += k * k;
  }
  sq[threadIdx.x] = aq; sk[threadIdx.x] = ak;
  __syncthreads();
  if (threadIdx.x < 32) {
    float q = 0.f, k = 0.f;
    for (int i = 0; i < 8; ++i) { q += sq[i * 32 + threadIdx.x]; k += sk[i * 32 + threadIdx.x]; }
    qcn[blockIdx.x * 32 + threadIdx.x] = fmaxf(sqrtf(q), 1e-12f);
    kcn[blockIdx.x * 32 + threadIdx.x] = fmaxf(sqrtf(k), 1e-12f);
  }
}

__global__ __launch_bounds__(256)
void ca_attn_kernel(const float* __restrict__ qkvv, const float* __restrict__ qcn,
                    const float* __restrict__ kcn, const float* __restrict__ temp,
                    float* __restrict__ attn)
{
  __shared__ float qch[32 * 33], kch[32 * 33];
  const int b = blockIdx.x >> 3, h = blockIdx.x & 7;
  const int w = threadIdx.x >> 5, e = threadIdx.x & 31;
  float acc[4] = {0.f, 0.f, 0.f, 0.f};
  for (int t0 = 0; t0 < NT; t0 += 32) {
    __syncthreads();
#pragma unroll
    for (int p = 0; p < 4; ++p) {
      const int tt = p * 8 + w;
      const size_t base = (size_t)(b * NT + t0 + tt) * 1024 + h * 32 + e;
      qch[e * 33 + tt] = qkvv[base];
      kch[e * 33 + tt] = qkvv[base + 256];
    }
    __syncthreads();
#pragma unroll 8
    for (int tt = 0; tt < 32; ++tt) {
      const float kv = kch[e * 33 + tt];
#pragma unroll
      for (int i = 0; i < 4; ++i) acc[i] += qch[(w + 8 * i) * 33 + tt] * kv;
    }
  }
  const float tmp = temp[h];
  const float kno = kcn[blockIdx.x * 32 + e];
#pragma unroll
  for (int i = 0; i < 4; ++i) {
    const int c = w + 8 * i;
    const float s = acc[i] / (qcn[blockIdx.x * 32 + c] * kno) * tmp;
    const float mx = wmax(s);
    const float ex = __expf(s - mx);
    const float sm = wsum(ex);
    attn[((size_t)blockIdx.x * 32 + c) * 32 + e] = ex / sm;
  }
}

__global__ __launch_bounds__(256)
void ca_out_kernel(const float* __restrict__ qkvv, const float* __restrict__ attn,
                   u16* __restrict__ xca)
{
  __shared__ float vsh[256];
  const int b = blockIdx.x / NT, t = blockIdx.x % NT;
  const int tid = threadIdx.x;
  vsh[tid] = qkvv[(size_t)(b * NT + t) * 1024 + 512 + tid];
  __syncthreads();
  const int h = tid >> 5, cc = tid & 31;
  const float* ar = attn + ((size_t)(b * NHc + h) * 32 + cc) * 32;
  const float* vr = vsh + h * 32;
  float s = 0.f;
#pragma unroll
  for (int e = 0; e < 32; ++e) s += ar[e] * vr[e];
  xca[(size_t)(b * NT + t) * 256 + tid] = f2bf(s);
}

// ---------------------------------------------------------------------------
// Residual: skip[b,c,t] = x[b,c,t] + gamma[c]*epa[b,t,c]; also bf16 act (b,t,c)
// ---------------------------------------------------------------------------
__global__ __launch_bounds__(256)
void residual_kernel(const float* __restrict__ x, const float* __restrict__ epa,
                     const float* __restrict__ gamma, float* __restrict__ skip,
                     u16* __restrict__ act)
{
  __shared__ float ld[256 * 33];
  const int b  = blockIdx.x / (NT / 32);
  const int t0 = (blockIdx.x % (NT / 32)) * 32;
  const int tid = threadIdx.x;
  for (int j = 0; j < 32; ++j)
    ld[tid * 33 + j] = epa[(size_t)(b * NT + t0 + j) * 256 + tid];
  __syncthreads();
  {
    const int j = tid & 31, cr = tid >> 5;
#pragma unroll
    for (int it = 0; it < 32; ++it) {
      const int c = it * 8 + cr;
      const size_t xi = (size_t)(b * 256 + c) * NT + t0 + j;
      const float sv = x[xi] + gamma[c] * ld[c * 33 + j];
      skip[xi] = sv;
      ld[c * 33 + j] = sv;
    }
  }
  __syncthreads();
  for (int j = 0; j < 32; ++j)
    act[(size_t)(b * NT + t0 + j) * 256 + tid] = f2bf(ld[tid * 33 + j]);
}

// ---------------------------------------------------------------------------
// BatchNorm stats / fold / apply
// ---------------------------------------------------------------------------
__global__ void zero512(float* p) { p[blockIdx.x * 256 + threadIdx.x] = 0.f; }

__global__ __launch_bounds__(256)
void stats_kernel(const float* __restrict__ co, float* __restrict__ stats)
{
  const int m0 = blockIdx.x * 128;
  const int c = threadIdx.x;
  float s = 0.f, s2 = 0.f;
  for (int r = 0; r < 128; ++r) {
    const float v = co[(size_t)(m0 + r) * 256 + c];
    s += v; s2 += v * v;
  }
  atomicAdd(&stats[c], s);
  atomicAdd(&stats[256 + c], s2);
}

__global__ void bnfold_kernel(float* stats, const float* w, const float* bb)
{
  const int c = threadIdx.x;
  const float inv = 1.f / (float)BT;
  const float mu = stats[c] * inv;
  const float var = stats[256 + c] * inv - mu * mu;
  const float sc = w[c] * rsqrtf(var + 1e-5f);
  stats[c] = sc;
  stats[256 + c] = bb[c] - mu * sc;
}

__global__ __launch_bounds__(256)
void bnapply_kernel(const float* __restrict__ co, const float* __restrict__ stats,
                    u16* __restrict__ act)
{
  const size_t i = (size_t)blockIdx.x * 256 + threadIdx.x;
  const int c = threadIdx.x;
  float v = co[i] * stats[c] + stats[256 + c];
  v = (v >= 0.f) ? v : 0.01f * v;
  act[i] = f2bf(v);
}

__global__ __launch_bounds__(256)
void bnapply2_kernel(const float* __restrict__ co, const float* __restrict__ stats,
                     const float* __restrict__ skip, u16* __restrict__ act)
{
  __shared__ float ld[256 * 33];
  const int b  = blockIdx.x / (NT / 32);
  const int t0 = (blockIdx.x % (NT / 32)) * 32;
  const int tid = threadIdx.x;
  for (int j = 0; j < 32; ++j)
    ld[tid * 33 + j] = co[(size_t)(b * NT + t0 + j) * 256 + tid];
  __syncthreads();
  {
    const int j = tid & 31, cr = tid >> 5;
#pragma unroll
    for (int it = 0; it < 32; ++it) {
      const int c = it * 8 + cr;
      float v = ld[c * 33 + j] * stats[c] + stats[256 + c] +
                skip[(size_t)(b * 256 + c) * NT + t0 + j];
      v = (v >= 0.f) ? v : 0.01f * v;
      ld[c * 33 + j] = v;
    }
  }
  __syncthreads();
  for (int j = 0; j < 32; ++j)
    act[(size_t)(b * NT + t0 + j) * 256 + tid] = f2bf(ld[tid * 33 + j]);
}

// ---------------------------------------------------------------------------
// Final: out[b,c,t] = skip[b,c,t] + o8[b,t,c]   (o8 already has conv8 bias)
// ---------------------------------------------------------------------------
__global__ __launch_bounds__(256)
void final_kernel(const float* __restrict__ skip, const float* __restrict__ o8,
                  float* __restrict__ out)
{
  __shared__ float ld[256 * 33];
  const int b  = blockIdx.x / (NT / 32);
  const int t0 = (blockIdx.x % (NT / 32)) * 32;
  const int tid = threadIdx.x;
  for (int j = 0; j < 32; ++j)
    ld[tid * 33 + j] = o8[(size_t)(b * NT + t0 + j) * 256 + tid];
  __syncthreads();
  const int j = tid & 31, cr = tid >> 5;
#pragma unroll
  for (int it = 0; it < 32; ++it) {
    const int c = it * 8 + cr;
    const size_t xi = (size_t)(b * 256 + c) * NT + t0 + j;
    out[xi] = skip[xi] + ld[c * 33 + j];
  }
}

// ---------------------------------------------------------------------------
// Weight conversion / reorder
// ---------------------------------------------------------------------------
__global__ void cvt_bf16_kernel(const float* __restrict__ in, u16* __restrict__ outp, int n)
{
  const int i = blockIdx.x * 256 + threadIdx.x;
  if (i < n) outp[i] = f2bf(in[i]);
}
__global__ void conv_reorder_kernel(const float* __restrict__ in, u16* __restrict__ outp)
{
  const int i = blockIdx.x * 256 + threadIdx.x;   // over 256*6912
  const int co = i / KCONV, rr = i % KCONV, tap = rr >> 8, ci = rr & 255;
  outp[i] = f2bf(in[(co * 256 + ci) * 27 + tap]);
}

// ---------------------------------------------------------------------------
// Host launcher
// ---------------------------------------------------------------------------
extern "C" void kernel_launch(void* const* d_in, const int* in_sizes, int n_in,
                              void* d_out, int out_size, void* d_ws, size_t ws_size,
                              hipStream_t stream)
{
  const float* x        = (const float*)d_in[0];
  const float* ln_w     = (const float*)d_in[1];
  const float* ln_b     = (const float*)d_in[2];
  const float* gamma    = (const float*)d_in[3];
  const float* w_qkvv   = (const float*)d_in[4];
  const float* temp1    = (const float*)d_in[5];
  const float* temp2    = (const float*)d_in[6];
  const float* rpb      = (const float*)d_in[7];
  const float* qembed   = (const float*)d_in[8];
  const float* w_op     = (const float*)d_in[9];
  const float* b_op     = (const float*)d_in[10];
  const float* w_op2    = (const float*)d_in[11];
  const float* b_op2    = (const float*)d_in[12];
  const float* conv1_w  = (const float*)d_in[13];
  const float* bn1_w    = (const float*)d_in[14];
  const float* bn1_b    = (const float*)d_in[15];
  const float* conv2_w  = (const float*)d_in[16];
  const float* bn2_w    = (const float*)d_in[17];
  const float* bn2_b    = (const float*)d_in[18];
  const float* conv8_w  = (const float*)d_in[19];
  const float* conv8_b  = (const float*)d_in[20];

  char* ws = (char*)d_ws;
  float* qkvv   = (float*)(ws + O_QKVV);
  u16*   xn     = (u16*)  (ws + O_XN);
  float* qs     = (float*)(ws + O_QS);
  float* kn     = (float*)(ws + O_KN);
  u16*   xsa    = (u16*)  (ws + O_XSA);
  u16*   xca    = (u16*)  (ws + O_XCA);
  float* epa    = (float*)(ws + O_EPA);
  float* skip   = (float*)(ws + O_SKIP);
  u16*   w1b    = (u16*)  (ws + O_W1);
  u16*   w2b    = (u16*)  (ws + O_W2);
  u16*   wqb    = (u16*)  (ws + O_WQ);
  u16*   wopb   = (u16*)  (ws + O_WOP);
  u16*   wop2b  = (u16*)  (ws + O_WOP2);
  u16*   w8b    = (u16*)  (ws + O_W8);
  float* stats  = (float*)(ws + O_STATS);
  float* qcn    = (float*)(ws + O_QCN);
  float* kcn    = (float*)(ws + O_KCN);
  float* attn   = (float*)(ws + O_ATTN);
  float* convout= (float*)(ws + O_CONVOUT);   // alias into qkvv region
  u16*   act    = (u16*)  (ws + O_ACT);       // alias into qkvv region
  float* o8     = (float*)(ws + O_O8);        // alias into qkvv region

  const int TTILES = Bc * (NT / 32);          // 1728 transpose-tile blocks

  // ---- weight prep (bf16 / reorder) ----
  cvt_bf16_kernel<<<(1024 * 256 + 255) / 256, 256, 0, stream>>>(w_qkvv, wqb, 1024 * 256);
  cvt_bf16_kernel<<<(128 * 256 + 255) / 256, 256, 0, stream>>>(w_op, wopb, 128 * 256);
  cvt_bf16_kernel<<<(128 * 256 + 255) / 256, 256, 0, stream>>>(w_op2, wop2b, 128 * 256);
  cvt_bf16_kernel<<<(256 * 256 + 255) / 256, 256, 0, stream>>>(conv8_w, w8b, 256 * 256);
  conv_reorder_kernel<<<(256 * KCONV) / 256, 256, 0, stream>>>(conv1_w, w1b);
  conv_reorder_kernel<<<(256 * KCONV) / 256, 256, 0, stream>>>(conv2_w, w2b);

  // ---- LN + QKVV projection (WMMA) ----
  ln_kernel<<<TTILES, 256, 0, stream>>>(x, ln_w, ln_b, xn);
  gemm_bf16<<<(BT / 128) * (1024 / 128), 256, 0, stream>>>(
      xn, wqb, qkvv, BT, 1024, 256, 1024, 0, nullptr);

  // ---- neighborhood self-attention ----
  qknorm_kernel<<<BT, 256, 0, stream>>>(qkvv, qembed, temp2, qs, kn);
  sa_kernel<<<BT, 256, 0, stream>>>(qs, kn, qkvv, rpb, xsa);

  // ---- channel (cross) attention ----
  ca_norm_kernel<<<Bc * NHc, 256, 0, stream>>>(qkvv, qcn, kcn);
  ca_attn_kernel<<<Bc * NHc, 256, 0, stream>>>(qkvv, qcn, kcn, temp1, attn);
  ca_out_kernel<<<BT, 256, 0, stream>>>(qkvv, attn, xca);

  // ---- output projections (fused concat via column offset) ----
  gemm_bf16<<<BT / 128, 256, 0, stream>>>(xsa, wopb,  epa, BT, 128, 256, 256, 0,   b_op);
  gemm_bf16<<<BT / 128, 256, 0, stream>>>(xca, wop2b, epa, BT, 128, 256, 256, 128, b_op2);

  // ---- residual -> skip (NCDHW f32) + conv input (bf16 channel-last) ----
  residual_kernel<<<TTILES, 256, 0, stream>>>(x, epa, gamma, skip, act);

  // ---- conv1 + BN1 + lrelu ----
  conv3d_bf16<<<(BT / 128) * 2, 256, 0, stream>>>(act, w1b, convout);
  zero512<<<2, 256, 0, stream>>>(stats);
  stats_kernel<<<BT / 128, 256, 0, stream>>>(convout, stats);
  bnfold_kernel<<<1, 256, 0, stream>>>(stats, bn1_w, bn1_b);
  bnapply_kernel<<<BT, 256, 0, stream>>>(convout, stats, act);

  // ---- conv2 + BN2 + skip + lrelu ----
  conv3d_bf16<<<(BT / 128) * 2, 256, 0, stream>>>(act, w2b, convout);
  zero512<<<2, 256, 0, stream>>>(stats);
  stats_kernel<<<BT / 128, 256, 0, stream>>>(convout, stats);
  bnfold_kernel<<<1, 256, 0, stream>>>(stats, bn2_w, bn2_b);
  bnapply2_kernel<<<TTILES, 256, 0, stream>>>(convout, stats, skip, act);

  // ---- conv8 (1x1x1 GEMM, WMMA) + final residual ----
  gemm_bf16<<<(BT / 128) * 2, 256, 0, stream>>>(act, w8b, o8, BT, 256, 256, 256, 0, conv8_b);
  final_kernel<<<TTILES, 256, 0, stream>>>(skip, o8, (float*)d_out);
}
